// HypergraphTemporalModel_4964982194496
// MI455X (gfx1250) — matrix operand order repaired and checked by reading, not compiled, not run on hardware
//
#include <hip/hip_runtime.h>
#include <hip/hip_bf16.h>

typedef __attribute__((ext_vector_type(16))) _Float16 v16h;
typedef __attribute__((ext_vector_type(8)))  _Float16 v8h;
typedef __attribute__((ext_vector_type(8)))  float    v8f;

#define N_NODES 8192
#define N_EDGES 2048
#define NNZV    262144
#define FDIM    127
#define DDIM    128
#define NHEAD   4
#define HDIM    32
#define CPW     4      // column tiles per wave in the WMMA GEMM

enum { ACT_NONE = 0, ACT_TANH = 1, ACT_RELU = 2 };

// Load a 16-half fragment from two 16B-aligned 8-half chunks.
__device__ __forceinline__ v16h load_frag(const _Float16* p0, const _Float16* p1)
{
    v8h lo = *(const v8h*)p0;
    v8h hi = *(const v8h*)p1;
    v16h r;
#pragma unroll
    for (int i = 0; i < 8; ++i) { r[i] = lo[i]; r[i + 8] = hi[i]; }
    return r;
}

// ---------------------------------------------------------------------------
// WMMA linear: Y = act(X[rows x K](f16) @ W[Dout x K](f16)^T + b0 + b1)
// One wave per (16-row, 64-col) macro tile; A-frag reused across 4 col tiles.
// K multiple of 32 (127-wide operands pre-padded). Fragments load as 16B v8h
// pairs -> global_load_b128, no guards, no cvt in the inner loop.
// A-frag (16x32): lane m=lane&15 is row; e 0..7 -> k=half*8+e, e 8..15 -> +16.
// B-frag (32x16): lane col n=lane&15; e 0..7 -> k=half*8+e, e 8..15 -> +16.
// C/D (16x16 f32): lane col n=lane&15; reg r -> row = r + 8*half.
// Epilogue writes f32 (Yf) and/or f16 (Yh).
// ---------------------------------------------------------------------------
__global__ __launch_bounds__(32)
void wmma_linear_kernel(const _Float16* __restrict__ X, int ldx, int K,
                        const _Float16* __restrict__ W,        // [Dout, K]
                        const float* __restrict__ b0,
                        const float* __restrict__ b1,
                        float* __restrict__ Yf,
                        _Float16* __restrict__ Yh,
                        int ldy, int act)
{
    const int lane = threadIdx.x;
    const int half = lane >> 4;
    const int m    = lane & 15;
    const int rowBase  = blockIdx.x * 16;
    const int colBase0 = blockIdx.y * (16 * CPW);

    v8f acc[CPW];
#pragma unroll
    for (int j = 0; j < CPW; ++j) acc[j] = (v8f){};

    const _Float16* xrow = X + (size_t)(rowBase + m) * ldx;
    const _Float16* wrow[CPW];
#pragma unroll
    for (int j = 0; j < CPW; ++j)
        wrow[j] = W + (size_t)(colBase0 + j * 16 + m) * K;

    for (int kBase = 0; kBase < K; kBase += 32) {
        v16h a = load_frag(xrow + kBase + half * 8,
                           xrow + kBase + 16 + half * 8);
#pragma unroll
        for (int j = 0; j < CPW; ++j) {
            v16h b = load_frag(wrow[j] + kBase + half * 8,
                               wrow[j] + kBase + 16 + half * 8);
            acc[j] = __builtin_amdgcn_wmma_f32_16x16x32_f16(false, a, false, b,
                                                            (short)0, acc[j],
                                                            false, false);
        }
    }

#pragma unroll
    for (int j = 0; j < CPW; ++j) {
        const int col = colBase0 + j * 16 + m;
        float bias = (b0 ? b0[col] : 0.0f) + (b1 ? b1[col] : 0.0f);
#pragma unroll
        for (int r = 0; r < 8; ++r) {
            float v = acc[j][r] + bias;
            if (act == ACT_TANH)      v = tanhf(v);
            else if (act == ACT_RELU) v = fmaxf(v, 0.0f);
            size_t idx = (size_t)(rowBase + half * 8 + r) * ldy + col;
            if (Yf) Yf[idx] = v;
            if (Yh) Yh[idx] = (_Float16)v;
        }
    }
}

// ---------------------------------------------------------------------------
// Packing / conversion kernels
// ---------------------------------------------------------------------------
__global__ void f32_to_f16_kernel(const float* __restrict__ src,
                                  _Float16* __restrict__ dst, int n)
{
    int i = blockIdx.x * blockDim.x + threadIdx.x;
    if (i < n) dst[i] = (_Float16)src[i];
}

__global__ void build_input1_kernel(const float* __restrict__ fea,
                                    const float* __restrict__ td,
                                    _Float16* __restrict__ inp1h)
{
    int t = blockIdx.x * blockDim.x + threadIdx.x;
    if (t >= N_NODES * DDIM) return;
    int n = t >> 7, d = t & 127;
    inp1h[t] = (_Float16)((d == 0) ? td[n] : fea[(size_t)n * FDIM + (d - 1)]);
}

__global__ void pad_fea_kernel(const float* __restrict__ fea,
                               _Float16* __restrict__ feah)
{
    int t = blockIdx.x * blockDim.x + threadIdx.x;
    if (t >= N_NODES * DDIM) return;
    int n = t >> 7, d = t & 127;
    feah[t] = (_Float16)((d < FDIM) ? fea[(size_t)n * FDIM + d] : 0.0f);
}

__global__ void pad_w_kernel(const float* __restrict__ Wsrc,   // [128,127]
                             _Float16* __restrict__ Wdst)       // [128,128] f16
{
    int t = blockIdx.x * blockDim.x + threadIdx.x;
    if (t >= DDIM * DDIM) return;
    int r = t >> 7, c = t & 127;
    Wdst[t] = (_Float16)((c < FDIM) ? Wsrc[(size_t)r * FDIM + c] : 0.0f);
}

// ---------------------------------------------------------------------------
// Hypergraph propagation (atomic segment sums, f32)
// ---------------------------------------------------------------------------
__global__ void degrees_kernel(const int* __restrict__ ni, const int* __restrict__ ei,
                               float* __restrict__ deg_n, float* __restrict__ deg_e)
{
    int i = blockIdx.x * blockDim.x + threadIdx.x;
    if (i >= NNZV) return;
    atomicAdd(&deg_n[ni[i]], 1.0f);
    atomicAdd(&deg_e[ei[i]], 1.0f);
}

__global__ void invert_kernel(float* __restrict__ v, int n)
{
    int i = blockIdx.x * blockDim.x + threadIdx.x;
    if (i >= n) return;
    float x = v[i];
    v[i] = (x > 0.0f) ? 1.0f / x : 0.0f;
}

__global__ void n2e_kernel(const float* __restrict__ x,
                           const int* __restrict__ ni, const int* __restrict__ ei,
                           float* __restrict__ efeat)
{
    int t = blockIdx.x * blockDim.x + threadIdx.x;
    int entry = t >> 7, d = t & 127;
    if (entry >= NNZV) return;
    atomicAdd(&efeat[(size_t)ei[entry] * DDIM + d], x[(size_t)ni[entry] * DDIM + d]);
}

__global__ void scale_edges_kernel(float* __restrict__ efeat,
                                   const float* __restrict__ b_inv)
{
    int t = blockIdx.x * blockDim.x + threadIdx.x;
    if (t >= N_EDGES * DDIM) return;
    efeat[t] *= b_inv[t >> 7];
}

__global__ void e2n_kernel(const float* __restrict__ efeat,
                           const int* __restrict__ ni, const int* __restrict__ ei,
                           float* __restrict__ agg)
{
    int t = blockIdx.x * blockDim.x + threadIdx.x;
    int entry = t >> 7, d = t & 127;
    if (entry >= NNZV) return;
    atomicAdd(&agg[(size_t)ni[entry] * DDIM + d], efeat[(size_t)ei[entry] * DDIM + d]);
}

// agg = agg * d_inv + bias ; optional f16 mirror for a following GEMM
__global__ void finalize_nodes_kernel(float* __restrict__ agg,
                                      const float* __restrict__ d_inv,
                                      const float* __restrict__ bias,
                                      _Float16* __restrict__ aggh)
{
    int t = blockIdx.x * blockDim.x + threadIdx.x;
    if (t >= N_NODES * DDIM) return;
    float v = agg[t] * d_inv[t >> 7] + bias[t & 127];
    agg[t] = v;
    if (aggh) aggh[t] = (_Float16)v;
}

// cat = [x[indice] (f32) | f (f16)] -> f16 [N, 256]
__global__ void concat_kernel(const float* __restrict__ x,
                              const _Float16* __restrict__ f,
                              const int* __restrict__ indice,
                              _Float16* __restrict__ cat)
{
    int t = blockIdx.x * blockDim.x + threadIdx.x;
    if (t >= N_NODES * DDIM) return;
    int n = t >> 7, d = t & 127;
    int src = indice[n];
    cat[(size_t)n * (2 * DDIM) + d]        = (_Float16)x[(size_t)src * DDIM + d];
    cat[(size_t)n * (2 * DDIM) + DDIM + d] = f[(size_t)n * DDIM + d];
}

// ---------------------------------------------------------------------------
// Flash attention, wave32 WMMA, transposed-score formulation, f16 in/out.
// One wave per (16-query block, head). qkv layout: [L, 3*D] f16, heads HDIM=32.
// S^T = K_blk @ Q^T so softmax stats are per-lane scalars (lane = query col).
// O^T = V^T @ P^T accumulated in two 16x16 f32 tiles (hd = 32).
// ---------------------------------------------------------------------------
__global__ __launch_bounds__(32)
void flash_attn_kernel(const _Float16* __restrict__ qkv, _Float16* __restrict__ out)
{
    const int L    = N_NODES;
    const int ld   = 3 * DDIM;
    const int lane = threadIdx.x;
    const int half = lane >> 4;
    const int q16  = lane & 15;
    const int qBase = blockIdx.x * 16;
    const int head  = blockIdx.y;

    const _Float16* Qp = qkv + head * HDIM;
    const _Float16* Kp = qkv + DDIM + head * HDIM;
    const _Float16* Vp = qkv + 2 * DDIM + head * HDIM;

    // Q as B-frag (K=hd=32): lane column q=q16, e 0..7 -> d = half*8+e ; e 8..15 -> 16+half*8+(e-8)
    v16h qb;
    {
        const _Float16* qrow = Qp + (size_t)(qBase + q16) * ld;
        qb = load_frag(qrow + half * 8, qrow + 16 + half * 8);
        const _Float16 hscale = (_Float16)0.17677669529663687f;   // 1/sqrt(32)
#pragma unroll
        for (int e = 0; e < 16; ++e) qb[e] *= hscale;
    }

    v8f o0 = {}, o1 = {};                 // O^T tiles: d 0-15, d 16-31
    float m_run = -1e30f, l_run = 0.0f;

    for (int kv = 0; kv < L; kv += 32) {
        const _Float16* krow0 = Kp + (size_t)(kv + q16) * ld;
        const _Float16* krow1 = Kp + (size_t)(kv + 16 + q16) * ld;
        if (kv + 32 < L) {
            __builtin_prefetch(krow0 + 32 * ld, 0, 0);
            __builtin_prefetch(krow1 + 32 * ld, 0, 0);
        }
        // K rows as A-frags: e 0..7 -> d = half*8+e ; e 8..15 -> 16+half*8+(e-8)
        v16h ka0 = load_frag(krow0 + half * 8, krow0 + 16 + half * 8);
        v16h ka1 = load_frag(krow1 + half * 8, krow1 + 16 + half * 8);

        v8f s0 = {}, s1 = {};
        s0 = __builtin_amdgcn_wmma_f32_16x16x32_f16(false, ka0, false, qb, (short)0, s0, false, false);
        s1 = __builtin_amdgcn_wmma_f32_16x16x32_f16(false, ka1, false, qb, (short)0, s1, false, false);

        // lane owns column q; regs r -> kv-local rows r+8*half (per tile)
        float bm = -1e30f;
#pragma unroll
        for (int r = 0; r < 8; ++r) bm = fmaxf(bm, fmaxf(s0[r], s1[r]));
        bm = fmaxf(bm, __shfl_xor(bm, 16));
        float m_new = fmaxf(m_run, bm);
        float corr  = __expf(m_run - m_new);

        float p0[8], p1[8], bs = 0.0f;
#pragma unroll
        for (int r = 0; r < 8; ++r) {
            p0[r] = __expf(s0[r] - m_new);
            p1[r] = __expf(s1[r] - m_new);
            bs += p0[r] + p1[r];
        }
        bs += __shfl_xor(bs, 16);
        l_run = l_run * corr + bs;
        m_run = m_new;
#pragma unroll
        for (int r = 0; r < 8; ++r) { o0[r] *= corr; o1[r] *= corr; }

        // Exchange halves to build P^T B-frag (K=32 over both S^T tiles)
        float x0[8], x1[8];
#pragma unroll
        for (int r = 0; r < 8; ++r) {
            x0[r] = __shfl_xor(p0[r], 16);
            x1[r] = __shfl_xor(p1[r], 16);
        }
        v16h pb;
#pragma unroll
        for (int e = 0; e < 16; ++e) {
            float v;
            if (half == 0) v = (e < 8) ? p0[e] : x0[e - 8];
            else           v = (e < 8) ? x1[e] : p1[e - 8];
            pb[e] = (_Float16)v;
        }

        // V^T as A-frags: m=lane&15 is d-local, elem e -> kv-local half*16+e
        v16h va0, va1;
#pragma unroll
        for (int e = 0; e < 16; ++e) {
            int kvl = half * 16 + e;
            const _Float16* vrow = Vp + (size_t)(kv + kvl) * ld;
            va0[e] = vrow[q16];
            va1[e] = vrow[16 + q16];
        }
        o0 = __builtin_amdgcn_wmma_f32_16x16x32_f16(false, va0, false, pb, (short)0, o0, false, false);
        o1 = __builtin_amdgcn_wmma_f32_16x16x32_f16(false, va1, false, pb, (short)0, o1, false, false);
    }

    float rinv = (l_run > 0.0f) ? 1.0f / l_run : 0.0f;
    _Float16* orow = out + (size_t)(qBase + q16) * DDIM + head * HDIM;
#pragma unroll
    for (int r = 0; r < 8; ++r) {
        orow[half * 8 + r]      = (_Float16)(o0[r] * rinv);
        orow[16 + half * 8 + r] = (_Float16)(o1[r] * rinv);
    }
}

// ---------------------------------------------------------------------------
// Final head: logits = h @ Wc2^T + bc2 (Dout=2), then row softmax. (f32)
// ---------------------------------------------------------------------------
__global__ void head_kernel(const float* __restrict__ h,
                            const float* __restrict__ Wc2,
                            const float* __restrict__ bc2,
                            float* __restrict__ out)
{
    int n = blockIdx.x * blockDim.x + threadIdx.x;
    if (n >= N_NODES) return;
    const float* hr = h + (size_t)n * DDIM;
    float z0 = bc2[0], z1 = bc2[1];
    for (int k = 0; k < DDIM; ++k) {
        float hv = hr[k];
        z0 += hv * Wc2[k];
        z1 += hv * Wc2[DDIM + k];
    }
    float mx = fmaxf(z0, z1);
    float e0 = __expf(z0 - mx), e1 = __expf(z1 - mx);
    float inv = 1.0f / (e0 + e1);
    out[(size_t)n * 2]     = e0 * inv;
    out[(size_t)n * 2 + 1] = e1 * inv;
}

// ---------------------------------------------------------------------------
extern "C" void kernel_launch(void* const* d_in, const int* in_sizes, int n_in,
                              void* d_out, int out_size, void* d_ws, size_t ws_size,
                              hipStream_t stream)
{
    const float* fea   = (const float*)d_in[0];
    const float* td    = (const float*)d_in[1];
    const int*   nidx  = (const int*)d_in[2];          // hyper_index row 0
    const int*   eidx  = (const int*)d_in[2] + NNZV;   // hyper_index row 1
    const int*   ind   = (const int*)d_in[3];
    const float* W1    = (const float*)d_in[5];
    const float* b1    = (const float*)d_in[6];
    const float* W2    = (const float*)d_in[7];
    const float* b2    = (const float*)d_in[8];
    const float* Wl1   = (const float*)d_in[9];
    const float* bl1   = (const float*)d_in[10];
    const float* Wih   = (const float*)d_in[11];
    const float* bih   = (const float*)d_in[12];
    const float* bhh   = (const float*)d_in[14];
    const float* Wl    = (const float*)d_in[15];
    const float* bl    = (const float*)d_in[16];
    const float* ipw   = (const float*)d_in[17];
    const float* ipb   = (const float*)d_in[18];
    const float* opw   = (const float*)d_in[19];
    const float* opb   = (const float*)d_in[20];
    const float* Wc1   = (const float*)d_in[21];
    const float* bc1   = (const float*)d_in[22];
    const float* Wc2   = (const float*)d_in[23];
    const float* bc2   = (const float*)d_in[24];
    float* out = (float*)d_out;

    const size_t ND = (size_t)N_NODES * DDIM;

    // ---- f32 workspace region ----
    float* wsf = (float*)d_ws;
    size_t off = 0;
    float* lin   = wsf + off; off += ND;                 // conv lin (scatter src)
    float* lin2  = wsf + off; off += ND;
    float* efeat = wsf + off; off += (size_t)N_EDGES * DDIM;
    float* agg1  = wsf + off; off += ND;                 // conv1 out
    float* agg2  = wsf + off; off += ND;                 // conv2 out
    float* hbuf  = wsf + off; off += ND;                 // classifier hidden
    float* deg_n = wsf + off; off += N_NODES;
    float* deg_e = wsf + off; off += N_EDGES;
    off = (off + 7) & ~(size_t)7;                        // 16B align for f16 region

    // ---- f16 workspace region ----
    _Float16* wsh = (_Float16*)(wsf + off);
    size_t ho = 0;
    _Float16* inp1h  = wsh + ho; ho += ND;
    _Float16* feah   = wsh + ho; ho += ND;
    _Float16* agg1h  = wsh + ho; ho += ND;               // conv1 out (f16 mirror)
    _Float16* f1h    = wsh + ho; ho += ND;               // liner1 out
    _Float16* fbufh  = wsh + ho; ho += ND;               // tanh RNN out
    _Float16* cath   = wsh + ho; ho += 2 * ND;           // [N,256]
    _Float16* inph   = wsh + ho; ho += ND;               // fuse out
    _Float16* qkvh   = wsh + ho; ho += 3 * ND;           // [N,384]
    _Float16* attnh  = wsh + ho; ho += ND;
    _Float16* sharedh= wsh + ho; ho += ND;
    _Float16* W1h    = wsh + ho; ho += (size_t)DDIM * DDIM;
    _Float16* W2h    = wsh + ho; ho += (size_t)DDIM * DDIM;
    _Float16* Wl1h   = wsh + ho; ho += (size_t)DDIM * DDIM;
    _Float16* Wihh   = wsh + ho; ho += (size_t)DDIM * DDIM;
    _Float16* Wlh    = wsh + ho; ho += (size_t)DDIM * 2 * DDIM;
    _Float16* ipwh   = wsh + ho; ho += (size_t)3 * DDIM * DDIM;
    _Float16* opwh   = wsh + ho; ho += (size_t)DDIM * DDIM;
    _Float16* Wc1h   = wsh + ho; ho += (size_t)DDIM * DDIM;
    (void)ws_size; (void)n_in; (void)in_sizes; (void)out_size;

    const int TB = 256;
    dim3 gND((N_NODES * DDIM + TB - 1) / TB);
    dim3 gNNZ(((size_t)NNZV * DDIM + TB - 1) / TB);
    dim3 gED((N_EDGES * DDIM + TB - 1) / TB);
    dim3 gW((DDIM * DDIM + TB - 1) / TB);

    // Zero atomic targets (graph-capture-safe async memsets)
    hipMemsetAsync(efeat, 0, (size_t)N_EDGES * DDIM * sizeof(float), stream);
    hipMemsetAsync(agg1,  0, ND * sizeof(float), stream);
    hipMemsetAsync(agg2,  0, ND * sizeof(float), stream);
    hipMemsetAsync(deg_n, 0, N_NODES * sizeof(float), stream);
    hipMemsetAsync(deg_e, 0, N_EDGES * sizeof(float), stream);

    // Degrees and inverses
    degrees_kernel<<<(NNZV + TB - 1) / TB, TB, 0, stream>>>(nidx, eidx, deg_n, deg_e);
    invert_kernel<<<(N_NODES + TB - 1) / TB, TB, 0, stream>>>(deg_n, N_NODES);
    invert_kernel<<<(N_EDGES + TB - 1) / TB, TB, 0, stream>>>(deg_e, N_EDGES);

    // Packing: input1 / padded fea to f16; weights to f16
    build_input1_kernel<<<gND, TB, 0, stream>>>(fea, td, inp1h);
    pad_fea_kernel<<<gND, TB, 0, stream>>>(fea, feah);
    pad_w_kernel<<<gW, TB, 0, stream>>>(Wl1, Wl1h);
    f32_to_f16_kernel<<<gW, TB, 0, stream>>>(W1, W1h, DDIM * DDIM);
    f32_to_f16_kernel<<<gW, TB, 0, stream>>>(W2, W2h, DDIM * DDIM);
    f32_to_f16_kernel<<<gW, TB, 0, stream>>>(Wih, Wihh, DDIM * DDIM);
    f32_to_f16_kernel<<<(2 * DDIM * DDIM + TB - 1) / TB, TB, 0, stream>>>(Wl, Wlh, 2 * DDIM * DDIM);
    f32_to_f16_kernel<<<(3 * DDIM * DDIM + TB - 1) / TB, TB, 0, stream>>>(ipw, ipwh, 3 * DDIM * DDIM);
    f32_to_f16_kernel<<<gW, TB, 0, stream>>>(opw, opwh, DDIM * DDIM);
    f32_to_f16_kernel<<<gW, TB, 0, stream>>>(Wc1, Wc1h, DDIM * DDIM);

    dim3 wmmaBlk(32);
    dim3 gLinD(N_NODES / 16, DDIM / (16 * CPW));

    // ---- HypergraphConv 1 ----
    wmma_linear_kernel<<<gLinD, wmmaBlk, 0, stream>>>(inp1h, DDIM, DDIM, W1h, nullptr, nullptr, lin, nullptr, DDIM, ACT_NONE);
    n2e_kernel<<<gNNZ, TB, 0, stream>>>(lin, nidx, eidx, efeat);
    scale_edges_kernel<<<gED, TB, 0, stream>>>(efeat, deg_e);
    e2n_kernel<<<gNNZ, TB, 0, stream>>>(efeat, nidx, eidx, agg1);
    finalize_nodes_kernel<<<gND, TB, 0, stream>>>(agg1, deg_n, b1, agg1h);

    // ---- HypergraphConv 2 (l == 2) ----
    hipMemsetAsync(efeat, 0, (size_t)N_EDGES * DDIM * sizeof(float), stream);
    wmma_linear_kernel<<<gLinD, wmmaBlk, 0, stream>>>(agg1h, DDIM, DDIM, W2h, nullptr, nullptr, lin2, nullptr, DDIM, ACT_NONE);
    n2e_kernel<<<gNNZ, TB, 0, stream>>>(lin2, nidx, eidx, efeat);
    scale_edges_kernel<<<gED, TB, 0, stream>>>(efeat, deg_e);
    e2n_kernel<<<gNNZ, TB, 0, stream>>>(efeat, nidx, eidx, agg2);
    finalize_nodes_kernel<<<gND, TB, 0, stream>>>(agg2, deg_n, b2, nullptr);

    // ---- Feature path: f = tanh((fea @ Wl1^T + bl1) @ Wih^T + bih + bhh) ----
    wmma_linear_kernel<<<gLinD, wmmaBlk, 0, stream>>>(feah, DDIM, DDIM, Wl1h, bl1, nullptr, nullptr, f1h, DDIM, ACT_NONE);
    wmma_linear_kernel<<<gLinD, wmmaBlk, 0, stream>>>(f1h, DDIM, DDIM, Wihh, bih, bhh, nullptr, fbufh, DDIM, ACT_TANH);

    // ---- Fuse: inp = [x[indice] | f] @ Wl^T + bl ----
    concat_kernel<<<gND, TB, 0, stream>>>(agg2, fbufh, ind, cath);
    wmma_linear_kernel<<<gLinD, wmmaBlk, 0, stream>>>(cath, 2 * DDIM, 2 * DDIM, Wlh, bl, nullptr, nullptr, inph, DDIM, ACT_NONE);

    // ---- MHA: QKV projection, flash attention, out-proj ----
    dim3 gQKV(N_NODES / 16, (3 * DDIM) / (16 * CPW));
    wmma_linear_kernel<<<gQKV, wmmaBlk, 0, stream>>>(inph, DDIM, DDIM, ipwh, ipb, nullptr, nullptr, qkvh, 3 * DDIM, ACT_NONE);
    dim3 gFA(N_NODES / 16, NHEAD);
    flash_attn_kernel<<<gFA, wmmaBlk, 0, stream>>>(qkvh, attnh);
    wmma_linear_kernel<<<gLinD, wmmaBlk, 0, stream>>>(attnh, DDIM, DDIM, opwh, opb, nullptr, nullptr, sharedh, DDIM, ACT_NONE);

    // ---- Classifier ----
    wmma_linear_kernel<<<gLinD, wmmaBlk, 0, stream>>>(sharedh, DDIM, DDIM, Wc1h, bc1, nullptr, hbuf, nullptr, DDIM, ACT_RELU);
    head_kernel<<<(N_NODES + TB - 1) / TB, TB, 0, stream>>>(hbuf, Wc2, bc2, out);
}
